// ReadPath_83820581749418
// MI455X (gfx1250) — matrix operand order repaired and checked by reading, not compiled, not run on hardware
//
#include <hip/hip_runtime.h>

// ---------------------------------------------------------------------------
// Problem constants (from reference)
// ---------------------------------------------------------------------------
#define BT      16384      // B*T
#define HDIM    2048       // H
#define DDIM    256        // D (belief dim)
#define NHEAD   4
#define NDQ     1024       // NH*D
#define NBEL    65536      // n_beliefs
#define NGOAL   16
#define TOPK    32
#define NEG_INF_F (-1.0e30f)
#define EPS_F   1e-8f

typedef unsigned int   uint32;
typedef __attribute__((ext_vector_type(4)))  int    v4i;
typedef __attribute__((ext_vector_type(16))) __bf16 v16bf;
typedef __attribute__((ext_vector_type(8)))  float  v8f;

union FragBF { uint4 q[2]; v16bf f; };

#define AS1 __attribute__((address_space(1)))
#define AS3 __attribute__((address_space(3)))

// Async global->LDS path (gfx1250). Guarded so either toolchain compiles.
#if __has_builtin(__builtin_amdgcn_global_load_async_to_lds_b128)
#define HAVE_ASYNC_LDS 1
#endif

#if __has_builtin(__builtin_amdgcn_s_wait_asynccnt)
#define WAIT_ASYNC(n) __builtin_amdgcn_s_wait_asynccnt((short)(n))
#else
#define WAIT_ASYNC(n) asm volatile("s_wait_asynccnt %0" ::"i"(n) : "memory")
#endif

// f32 -> bf16 round-to-nearest-even, raw bits
static __device__ __forceinline__ unsigned short f2bf(float f) {
    unsigned int u = __float_as_uint(f);
    unsigned int r = 0x7FFFu + ((u >> 16) & 1u);
    return (unsigned short)((u + r) >> 16);
}
static __device__ __forceinline__ uint32 pack2(float lo, float hi) {
    return (uint32)f2bf(lo) | ((uint32)f2bf(hi) << 16);
}
static __device__ __forceinline__ float waveSum(float v) {
    #pragma unroll
    for (int o = 16; o > 0; o >>= 1) v += __shfl_xor(v, o);
    return v;
}

// ---------------------------------------------------------------------------
// f32 -> bf16 bulk conversion: each thread converts 8 floats -> one uint4
// ---------------------------------------------------------------------------
__global__ __launch_bounds__(256) void cvt_bf16_kernel(const float* __restrict__ s,
                                                       uint4* __restrict__ d) {
    size_t i = (size_t)blockIdx.x * 256 + threadIdx.x;
    const float4* p = (const float4*)(s + i * 8);
    float4 a = p[0], b = p[1];
    uint4 o;
    o.x = pack2(a.x, a.y); o.y = pack2(a.z, a.w);
    o.z = pack2(b.x, b.y); o.w = pack2(b.z, b.w);
    d[i] = o;
}

// ---------------------------------------------------------------------------
// Stage 1: mean of hidden over B*T rows (two-stage, deterministic)
// ---------------------------------------------------------------------------
__global__ void mean_stage1(const float* __restrict__ hidden, float* __restrict__ part) {
    int chunk = blockIdx.x >> 3;
    int h     = (blockIdx.x & 7) * 256 + threadIdx.x;
    const float* p = hidden + (size_t)chunk * 512u * HDIM + h;
    float s = 0.f;
    for (int r = 0; r < 512; ++r) s += p[(size_t)r * HDIM];
    part[chunk * HDIM + h] = s;
}
__global__ void mean_stage2(const float* __restrict__ part, float* __restrict__ meanq) {
    int h = blockIdx.x * 256 + threadIdx.x;
    float s = 0.f;
    for (int c = 0; c < 32; ++c) s += part[c * HDIM + h];
    meanq[h] = s * (1.0f / (float)BT);
}

// ---------------------------------------------------------------------------
// Stage 2: rough_query = mean over heads of (Wq @ mean_query)
// ---------------------------------------------------------------------------
__global__ void rqfull_kernel(const float* __restrict__ Wq, const float* __restrict__ meanq,
                              float* __restrict__ rqf) {
    int i = blockIdx.x * 256 + threadIdx.x;
    const float* wrow = Wq + (size_t)i * HDIM;
    float s = 0.f;
    for (int h = 0; h < HDIM; ++h) s += wrow[h] * meanq[h];
    rqf[i] = s;
}
__global__ void roughq_kernel(const float* __restrict__ rqf, float* __restrict__ rq) {
    int d = threadIdx.x;
    rq[d] = 0.25f * (rqf[d] + rqf[DDIM + d] + rqf[2 * DDIM + d] + rqf[3 * DDIM + d]);
}

// ---------------------------------------------------------------------------
// Stage 3: rough scores, one wave32 per belief
// ---------------------------------------------------------------------------
__global__ __launch_bounds__(256) void rough_scores_kernel(
    const float* __restrict__ beliefs, const unsigned char* __restrict__ mask,
    const float* __restrict__ rq, float* __restrict__ scores) {
    int tid = threadIdx.x, lane = tid & 31, wave = tid >> 5;
    int b = blockIdx.x * 8 + wave;
    const float4* row = (const float4*)(beliefs + (size_t)b * DDIM + lane * 8);
    const float4* q   = (const float4*)(rq + lane * 8);
    float4 x0 = row[0], x1 = row[1], q0 = q[0], q1 = q[1];
    float dot = x0.x*q0.x + x0.y*q0.y + x0.z*q0.z + x0.w*q0.w
              + x1.x*q1.x + x1.y*q1.y + x1.z*q1.z + x1.w*q1.w;
    float n2  = x0.x*x0.x + x0.y*x0.y + x0.z*x0.z + x0.w*x0.w
              + x1.x*x1.x + x1.y*x1.y + x1.z*x1.z + x1.w*x1.w;
    dot = waveSum(dot);
    n2  = waveSum(n2);
    if (lane == 0) {
        float r = fmaxf(sqrtf(n2), EPS_F);
        scores[b] = mask[b] ? (dot / r) : NEG_INF_F;
    }
}

// ---------------------------------------------------------------------------
// Stage 4: iterative top-32 (single block; ties -> lower index like lax.top_k)
// ---------------------------------------------------------------------------
__global__ __launch_bounds__(1024) void topk_kernel(float* __restrict__ scores,
                                                    int* __restrict__ topk_idx) {
    __shared__ float sv[1024];
    __shared__ int   si[1024];
    int tid = threadIdx.x;
    for (int k = 0; k < TOPK; ++k) {
        float best = -3.4e38f; int bi = 0x7fffffff;
        for (int i = tid; i < NBEL; i += 1024) {
            float v = scores[i];
            if (v > best || (v == best && i < bi)) { best = v; bi = i; }
        }
        sv[tid] = best; si[tid] = bi;
        __syncthreads();
        for (int off = 512; off > 0; off >>= 1) {
            if (tid < off) {
                float v2 = sv[tid + off]; int i2 = si[tid + off];
                if (v2 > sv[tid] || (v2 == sv[tid] && i2 < si[tid])) { sv[tid] = v2; si[tid] = i2; }
            }
            __syncthreads();
        }
        if (tid == 0) { topk_idx[k] = si[0]; scores[si[0]] = NEG_INF_F; }
        __syncthreads();
    }
}

// ---------------------------------------------------------------------------
// Stage 5a: goal angle normalization + temperature (one wave per goal)
// ---------------------------------------------------------------------------
__global__ __launch_bounds__(512) void prep_goals(const float* __restrict__ ge,
                                                  const float* __restrict__ logt,
                                                  float* __restrict__ gang,
                                                  float* __restrict__ temp) {
    int tid = threadIdx.x, lane = tid & 31, g = tid >> 5;
    const float4* p = (const float4*)(ge + (size_t)g * DDIM + lane * 8);
    float4 a = p[0], b = p[1];
    float n2 = a.x*a.x + a.y*a.y + a.z*a.z + a.w*a.w
             + b.x*b.x + b.y*b.y + b.z*b.z + b.w*b.w;
    n2 = waveSum(n2);
    float inv = 1.0f / fmaxf(sqrtf(n2), EPS_F);
    float4* o = (float4*)(gang + (size_t)g * DDIM + lane * 8);
    o[0] = make_float4(a.x*inv, a.y*inv, a.z*inv, a.w*inv);
    o[1] = make_float4(b.x*inv, b.y*inv, b.z*inv, b.w*inv);
    if (tid < NHEAD) temp[tid] = fmaxf(expf(logt[tid]), 0.1f);
}

// ---------------------------------------------------------------------------
// Stage 5b: keys/values gather + goal bias (one block per selected belief)
// ---------------------------------------------------------------------------
__global__ __launch_bounds__(256) void prep_keys(const float* __restrict__ beliefs,
                                                 const int* __restrict__ topk_idx,
                                                 const float* __restrict__ gang,
                                                 const float* __restrict__ gpri,
                                                 float* __restrict__ keys,
                                                 float* __restrict__ values,
                                                 float* __restrict__ gbias) {
    __shared__ float red[256];
    __shared__ float kl[256];
    __shared__ float sg[NGOAL];
    int k = blockIdx.x, tid = threadIdx.x;
    int idx = topk_idx[k];
    float v = beliefs[(size_t)idx * DDIM + tid];
    red[tid] = v * v;
    __syncthreads();
    for (int off = 128; off > 0; off >>= 1) {
        if (tid < off) red[tid] += red[tid + off];
        __syncthreads();
    }
    float r = fmaxf(sqrtf(red[0]), EPS_F);
    float key = v / r;
    keys[k * DDIM + tid]   = key;
    values[k * DDIM + tid] = v;
    kl[tid] = key;
    __syncthreads();
    if (tid < NGOAL) {
        float dot = 0.f;
        for (int d = 0; d < DDIM; ++d) dot += kl[d] * gang[tid * DDIM + d];
        sg[tid] = dot * gpri[tid];
    }
    __syncthreads();
    if (tid == 0) {
        float m = sg[0];
        for (int g = 1; g < NGOAL; ++g) m = fmaxf(m, sg[g]);
        gbias[k] = m;
    }
}

// ---------------------------------------------------------------------------
// WMMA bf16 GEMM (NT): C[M,N] = A[M,K] * B[N,K]^T. A,B already bf16 in global.
// 64x64 macro-tile, 8 waves (4Mx2N), each wave 16x32 via v_wmma_f32_16x16x32_bf16.
// Async global->LDS double-buffered when toolchain exposes the builtin.
// ---------------------------------------------------------------------------
#define GTM 64
#define GTN 64
#define GKC 128
__global__ __launch_bounds__(256) void gemm_nt_bf16(const unsigned short* __restrict__ A,
                                                    const unsigned short* __restrict__ B,
                                                    float* __restrict__ C,
                                                    int M, int N, int Kdim) {
    __shared__ uint32 lA[2][GTM * (GKC / 2)];   // 2 x 16KB, row stride 64 uints
    __shared__ uint32 lB[2][GTN * (GKC / 2)];
    const int tid  = threadIdx.x;
    const int lane = tid & 31;
    const int wave = tid >> 5;
    const int wm = wave & 3;
    const int wn = wave >> 2;
    const int m0 = blockIdx.y * GTM;
    const int n0 = blockIdx.x * GTN;
    const int r  = lane & 15;
    const int hi = lane >> 4;
    v8f acc0 = {}; v8f acc1 = {};
    const int NC = Kdim / GKC;

#if defined(HAVE_ASYNC_LDS)
    // --- async double-buffered pipeline: 8 async b128 instructions / wave / chunk
    auto issue = [&](int chunk, int buf) {
        const int k0 = chunk * GKC;
        #pragma unroll
        for (int it = 0; it < 4; ++it) {
            int g = tid + it * 256;            // 1024 granules of 16B
            int row = g >> 4;
            int c16 = g & 15;
            const unsigned short* ga = A + (size_t)(m0 + row) * Kdim + k0 + c16 * 8;
            const unsigned short* gb = B + (size_t)(n0 + row) * Kdim + k0 + c16 * 8;
            __builtin_amdgcn_global_load_async_to_lds_b128(
                (AS1 v4i*)ga, (AS3 v4i*)&lA[buf][row * 64 + c16 * 4], 0, 0);
            __builtin_amdgcn_global_load_async_to_lds_b128(
                (AS1 v4i*)gb, (AS3 v4i*)&lB[buf][row * 64 + c16 * 4], 0, 0);
        }
    };
    issue(0, 0);
    for (int c = 0; c < NC; ++c) {
        const int cur = c & 1;
        if (c + 1 < NC) {
            issue(c + 1, cur ^ 1);
            WAIT_ASYNC(8);     // in-order completion: chunk c has landed
        } else {
            WAIT_ASYNC(0);
        }
        __syncthreads();
        #pragma unroll
        for (int ks = 0; ks < GKC / 32; ++ks) {
            FragBF a;
            const uint32* pa = &lA[cur][(wm * 16 + r) * 64 + ks * 16 + hi * 4];
            a.q[0] = *(const uint4*)(pa);
            a.q[1] = *(const uint4*)(pa + 8);
            FragBF b0, b1;
            const uint32* pb0 = &lB[cur][(wn * 32 + r) * 64 + ks * 16 + hi * 4];
            b0.q[0] = *(const uint4*)(pb0);
            b0.q[1] = *(const uint4*)(pb0 + 8);
            const uint32* pb1 = pb0 + 16 * 64;
            b1.q[0] = *(const uint4*)(pb1);
            b1.q[1] = *(const uint4*)(pb1 + 8);
            acc0 = __builtin_amdgcn_wmma_f32_16x16x32_bf16(false, a.f, false, b0.f,
                                                           (short)0, acc0, false, false);
            acc1 = __builtin_amdgcn_wmma_f32_16x16x32_bf16(false, a.f, false, b1.f,
                                                           (short)0, acc1, false, false);
        }
        __syncthreads();
    }
#else
    // --- fallback: synchronous bf16 tile copies through VGPRs
    for (int c = 0; c < NC; ++c) {
        const int k0 = c * GKC;
        __syncthreads();
        #pragma unroll
        for (int it = 0; it < 4; ++it) {
            int g = tid + it * 256;
            int row = g >> 4;
            int c16 = g & 15;
            *(uint4*)&lA[0][row * 64 + c16 * 4] =
                *(const uint4*)(A + (size_t)(m0 + row) * Kdim + k0 + c16 * 8);
            *(uint4*)&lB[0][row * 64 + c16 * 4] =
                *(const uint4*)(B + (size_t)(n0 + row) * Kdim + k0 + c16 * 8);
            if (k0 + GKC < Kdim) {
                __builtin_prefetch(A + (size_t)(m0 + row) * Kdim + k0 + GKC + c16 * 8, 0, 0);
                __builtin_prefetch(B + (size_t)(n0 + row) * Kdim + k0 + GKC + c16 * 8, 0, 0);
            }
        }
        __syncthreads();
        #pragma unroll
        for (int ks = 0; ks < GKC / 32; ++ks) {
            FragBF a;
            const uint32* pa = &lA[0][(wm * 16 + r) * 64 + ks * 16 + hi * 4];
            a.q[0] = *(const uint4*)(pa);
            a.q[1] = *(const uint4*)(pa + 8);
            FragBF b0, b1;
            const uint32* pb0 = &lB[0][(wn * 32 + r) * 64 + ks * 16 + hi * 4];
            b0.q[0] = *(const uint4*)(pb0);
            b0.q[1] = *(const uint4*)(pb0 + 8);
            const uint32* pb1 = pb0 + 16 * 64;
            b1.q[0] = *(const uint4*)(pb1);
            b1.q[1] = *(const uint4*)(pb1 + 8);
            acc0 = __builtin_amdgcn_wmma_f32_16x16x32_bf16(false, a.f, false, b0.f,
                                                           (short)0, acc0, false, false);
            acc1 = __builtin_amdgcn_wmma_f32_16x16x32_bf16(false, a.f, false, b1.f,
                                                           (short)0, acc1, false, false);
        }
    }
#endif

    // C layout: lanes 0-15 -> M=i, lanes 16-31 -> M=i+8; col = lane&15
    #pragma unroll
    for (int i = 0; i < 8; ++i) {
        int mOff = hi ? (i + 8) : i;
        size_t rowIdx = (size_t)(m0 + wm * 16 + mOff) * (size_t)N;
        C[rowIdx + n0 + wn * 32 + r]      = acc0[i];
        C[rowIdx + n0 + wn * 32 + 16 + r] = acc1[i];
    }
}

// ---------------------------------------------------------------------------
// Stage 7: fused per-row attention (one wave per bt row, lane = key index);
// writes retrieved directly as bf16 for GEMM2.
// ---------------------------------------------------------------------------
__global__ __launch_bounds__(256) void attn_kernel(const float* __restrict__ Q,
                                                   const float* __restrict__ keys,
                                                   const float* __restrict__ values,
                                                   const float* __restrict__ gbias,
                                                   const float* __restrict__ temp,
                                                   unsigned short* __restrict__ Rbf) {
    __shared__ float kl[TOPK][DDIM + 1];
    __shared__ float attn[8][TOPK];
    int tid = threadIdx.x, lane = tid & 31, wave = tid >> 5;
    for (int i = tid; i < TOPK * DDIM; i += 256) {
        kl[i >> 8][i & 255] = keys[i];
    }
    __syncthreads();
    size_t row = (size_t)blockIdx.x * 8 + wave;
    const float invS = 0.0625f;
    float gb = gbias[lane];
    for (int h = 0; h < NHEAD; ++h) {
        const float* q = Q + row * NDQ + h * DDIM;
        float th = temp[h];
        float s = 0.f;
        #pragma unroll 4
        for (int d = 0; d < DDIM; ++d) s += q[d] * kl[lane][d];
        s = s * invS * th + gb;
        float m = s;
        #pragma unroll
        for (int o = 16; o > 0; o >>= 1) m = fmaxf(m, __shfl_xor(m, o));
        float e = __expf(s - m);
        float sum = e;
        #pragma unroll
        for (int o = 16; o > 0; o >>= 1) sum += __shfl_xor(sum, o);
        attn[wave][lane] = e / sum;
        unsigned short* out = Rbf + row * NDQ + h * DDIM;
        #pragma unroll
        for (int j = 0; j < 8; ++j) {
            int d = lane + j * 32;
            float acc = 0.f;
            #pragma unroll 8
            for (int n = 0; n < TOPK; ++n) acc += attn[wave][n] * values[n * DDIM + d];
            out[d] = f2bf(acc);
        }
    }
}

// ---------------------------------------------------------------------------
// Host launcher
// ---------------------------------------------------------------------------
extern "C" void kernel_launch(void* const* d_in, const int* in_sizes, int n_in,
                              void* d_out, int out_size, void* d_ws, size_t ws_size,
                              hipStream_t stream) {
    const float*         hidden  = (const float*)d_in[0];
    const float*         beliefs = (const float*)d_in[1];
    const unsigned char* mask    = (const unsigned char*)d_in[2];   // jax bool -> 1 byte
    const float*         ge      = (const float*)d_in[3];
    const float*         gpri    = (const float*)d_in[4];
    const float*         Wq      = (const float*)d_in[5];
    const float*         Wo      = (const float*)d_in[6];
    const float*         logt    = (const float*)d_in[7];
    float*               out     = (float*)d_out;

    char* w = (char*)d_ws;
    constexpr size_t OFF_Q      = 0;                                    // 64 MB f32
    constexpr size_t OFF_HBF    = OFF_Q      + (size_t)BT * NDQ * 4;    // 64 MB bf16
    constexpr size_t OFF_RBF    = OFF_HBF    + (size_t)BT * HDIM * 2;   // 32 MB bf16
    constexpr size_t OFF_WQBF   = OFF_RBF    + (size_t)BT * NDQ * 2;    // 4 MB bf16
    constexpr size_t OFF_WOBF   = OFF_WQBF   + (size_t)NDQ * HDIM * 2;  // 4 MB bf16
    constexpr size_t OFF_PART   = OFF_WOBF   + (size_t)HDIM * NDQ * 2;
    constexpr size_t OFF_MEANQ  = OFF_PART   + 32 * HDIM * 4;
    constexpr size_t OFF_RQFULL = OFF_MEANQ  + HDIM * 4;
    constexpr size_t OFF_RQ     = OFF_RQFULL + NDQ * 4;
    constexpr size_t OFF_SCORES = OFF_RQ     + DDIM * 4;
    constexpr size_t OFF_TOPK   = OFF_SCORES + (size_t)NBEL * 4;
    constexpr size_t OFF_KEYS   = OFF_TOPK   + 128;
    constexpr size_t OFF_VALS   = OFF_KEYS   + TOPK * DDIM * 4;
    constexpr size_t OFF_GANG   = OFF_VALS   + TOPK * DDIM * 4;
    constexpr size_t OFF_GBIAS  = OFF_GANG   + NGOAL * DDIM * 4;
    constexpr size_t OFF_TEMP   = OFF_GBIAS  + 128;

    float*          Qb     = (float*)(w + OFF_Q);
    unsigned short* hbf    = (unsigned short*)(w + OFF_HBF);
    unsigned short* rbf    = (unsigned short*)(w + OFF_RBF);
    unsigned short* wqbf   = (unsigned short*)(w + OFF_WQBF);
    unsigned short* wobf   = (unsigned short*)(w + OFF_WOBF);
    float*          part   = (float*)(w + OFF_PART);
    float*          meanq  = (float*)(w + OFF_MEANQ);
    float*          rqf    = (float*)(w + OFF_RQFULL);
    float*          rq     = (float*)(w + OFF_RQ);
    float*          scores = (float*)(w + OFF_SCORES);
    int*            tkidx  = (int*)  (w + OFF_TOPK);
    float*          keys   = (float*)(w + OFF_KEYS);
    float*          vals   = (float*)(w + OFF_VALS);
    float*          gang   = (float*)(w + OFF_GANG);
    float*          gbias  = (float*)(w + OFF_GBIAS);
    float*          temp   = (float*)(w + OFF_TEMP);

    // bf16 operand conversions (one streaming pass each)
    cvt_bf16_kernel<<<(BT * HDIM) / (256 * 8), 256, 0, stream>>>(hidden, (uint4*)hbf);
    cvt_bf16_kernel<<<(NDQ * HDIM) / (256 * 8), 256, 0, stream>>>(Wq, (uint4*)wqbf);
    cvt_bf16_kernel<<<(HDIM * NDQ) / (256 * 8), 256, 0, stream>>>(Wo, (uint4*)wobf);

    mean_stage1<<<256, 256, 0, stream>>>(hidden, part);
    mean_stage2<<<8, 256, 0, stream>>>(part, meanq);
    rqfull_kernel<<<4, 256, 0, stream>>>(Wq, meanq, rqf);
    roughq_kernel<<<1, 256, 0, stream>>>(rqf, rq);
    rough_scores_kernel<<<NBEL / 8, 256, 0, stream>>>(beliefs, mask, rq, scores);
    topk_kernel<<<1, 1024, 0, stream>>>(scores, tkidx);
    prep_goals<<<1, 512, 0, stream>>>(ge, logt, gang, temp);
    prep_keys<<<TOPK, 256, 0, stream>>>(beliefs, tkidx, gang, gpri, keys, vals, gbias);

    // GEMM1: Q[BT, NDQ] = hidden_bf16 @ Wq_bf16^T
    dim3 g1(NDQ / GTN, BT / GTM);
    gemm_nt_bf16<<<g1, 256, 0, stream>>>(hbf, wqbf, Qb, BT, NDQ, HDIM);

    attn_kernel<<<BT / 8, 256, 0, stream>>>(Qb, keys, vals, gbias, temp, rbf);

    // GEMM2: out[BT, HDIM] = R_bf16 @ Wo_bf16^T
    dim3 g2(HDIM / GTN, BT / GTM);
    gemm_nt_bf16<<<g2, 256, 0, stream>>>(rbf, wobf, out, BT, HDIM, NDQ);

    (void)in_sizes; (void)n_in; (void)out_size; (void)ws_size;
}
